// SlotAttention_6390911337138
// MI455X (gfx1250) — matrix operand order repaired and checked by reading, not compiled, not run on hardware
//
#include <hip/hip_runtime.h>

// ---------------------------------------------------------------------------
// Slot Attention forward for MI455X (gfx1250, wave32, WMMA bf16 16x16x32).
// HBM-bound problem (~0.9 GB traffic vs ~37 GFLOP): bf16 k/v fragments halve
// the iteration-loop traffic; all accumulation in fp32 via v_wmma_f32_*.
// Input staging uses the Tensor Data Mover (tensor_load_to_lds + TENSORcnt).
// ---------------------------------------------------------------------------

constexpr int kB  = 64;
constexpr int kN  = 4096;
constexpr int kDF = 256;
constexpr int kDS = 128;
constexpr int kS  = 8;
constexpr int kMH = 256;

#define SCALE_F  0.08838834764831845f   // 128^-0.5
#define LN_EPS_F 1e-5f
#define EPS_F    1e-8f

typedef __attribute__((ext_vector_type(16))) __bf16 v16bf;
typedef __attribute__((ext_vector_type(8)))  float  v8f;

// TDM descriptor groups (6-arg builtin form on this toolchain)
typedef __attribute__((ext_vector_type(4))) unsigned int tdm_g0_t;
typedef __attribute__((ext_vector_type(8))) int          tdm_g1_t;
typedef __attribute__((ext_vector_type(4))) int          tdm_g2_t;
typedef __attribute__((ext_vector_type(8))) int          tdm_g4_t;

static __device__ __forceinline__ unsigned short f2bf(float f) {
  unsigned u = __float_as_uint(f);
  u += 0x7FFFu + ((u >> 16) & 1u);          // round-to-nearest-even
  return (unsigned short)(u >> 16);
}
static __device__ __forceinline__ unsigned pack2(float a, float b) {
  return (unsigned)f2bf(a) | ((unsigned)f2bf(b) << 16);
}

// ---------------------------------------------------------------------------
// Pack Wk/Wv (DFxDS fp32, row-major) into bf16 WMMA B-fragments.
// Layout: [mat(2)][ctile(8)][ktile(8)][lane(32)][vgpr(8)] as u32 words.
// B-frag (32x16 bf16): lane L: n = L%16, K = (L/16)*8 + (j<4 ? 2j : 16+2(j-4)) {,+1}.
// ---------------------------------------------------------------------------
__global__ void prep_wfrags_kernel(const float* __restrict__ Wk,
                                   const float* __restrict__ Wv,
                                   unsigned* __restrict__ wf) {
  const int idx = blockIdx.x * blockDim.x + threadIdx.x;   // 0..32767
  const int j    = idx & 7;
  const int lane = (idx >> 3) & 31;
  const int kt   = (idx >> 8) & 7;
  const int ct   = (idx >> 11) & 7;
  const int mat  = (idx >> 14) & 1;
  const float* W = mat ? Wv : Wk;
  const int n = ct * 16 + (lane & 15);
  const int k = kt * 32 + (lane >> 4) * 8 + ((j < 4) ? 2 * j : 16 + 2 * (j - 4));
  wf[idx] = pack2(W[k * kDS + n], W[(k + 1) * kDS + n]);
}

// ---------------------------------------------------------------------------
// Fused LayerNorm(inputs) + k/v projection, emitting fragment layouts:
//   k_frags: [b][ttile(256)][kt(4)][lane][8]  (K = d,   N = token)  for dots
//   v_frags: [b][chunk(128)][dt(8)][lane][8]  (K = tok, N = d)      for updates
// Block = 256 thr (8 waves), 32 tokens per block. Input tile (32x256 fp32,
// 32 KB) is DMA'd into LDS by the Tensor Data Mover, then LN runs from LDS.
// ---------------------------------------------------------------------------
__global__ void ln_kv_kernel(const float* __restrict__ inputs,
                             const float* __restrict__ g_in,
                             const float* __restrict__ b_in,
                             const unsigned* __restrict__ w_frags,
                             unsigned* __restrict__ k_frags,
                             unsigned* __restrict__ v_frags) {
  __shared__ float x_lds[32][kDF];                  // raw input tile (TDM), 32 KB
  __shared__ unsigned short a_lds[32][kDF];         // LN'd bf16, 16 KB
  __shared__ unsigned short kv_stage[2][32][kDS];   // 16 KB
  const int t = threadIdx.x, wave = t >> 5, lane = t & 31;
  const int b = blockIdx.x >> 7;
  const int chunk = blockIdx.x & 127;
  const int tok0 = chunk << 5;

  // ---- Phase 0: TDM DMA of the 32x256 fp32 tile into LDS (one wave issues) --
  if (wave == 0) {
    const unsigned long long gaddr =
        (unsigned long long)(uintptr_t)(inputs + (size_t)(b * kN + tok0) * kDF);
    const unsigned lds_off = (unsigned)(uintptr_t)(&x_lds[0][0]);  // flat[31:0] = LDS offset
    tdm_g0_t g0;
    g0[0] = (unsigned)__builtin_amdgcn_readfirstlane(1u);          // count=1 (valid user D#)
    g0[1] = (unsigned)__builtin_amdgcn_readfirstlane(lds_off);     // lds_addr
    g0[2] = (unsigned)__builtin_amdgcn_readfirstlane((unsigned)gaddr);
    g0[3] = (unsigned)__builtin_amdgcn_readfirstlane(
                (unsigned)(gaddr >> 32) | 0x80000000u);            // addr[56:32] | type=2
    tdm_g1_t g1;
    g1[0] = (int)(2u << 16);                      // workgroup_mask=0, data_size=2 (4B)
    g1[1] = (int)(((unsigned)kDF & 0xFFFFu) << 16);          // tensor_dim0[15:0]
    g1[2] = (int)(((unsigned)kDF >> 16) | (32u << 16));      // td0[31:16] | tensor_dim1=32
    g1[3] = (int)((unsigned)kDF << 16);                      // td1[31:16]=0 | tile_dim0=256
    g1[4] = 32;                                   // tile_dim1=32, tile_dim2=0
    g1[5] = kDF;                                  // tensor_dim0_stride = 256 elements
    g1[6] = 0;
    g1[7] = 0;
    tdm_g2_t z4 = {0, 0, 0, 0};                   // groups 2/3 unused (2-D tensor)
    tdm_g4_t z8 = {0, 0, 0, 0, 0, 0, 0, 0};
    __builtin_amdgcn_tensor_load_to_lds(g0, g1, z4, z4, z8, 0);
  }
  // Warm WGP cache with the weight fragments (128 KB, L2-resident) meanwhile.
#pragma unroll
  for (int i = 0; i < 4; ++i)
    __builtin_prefetch((const char*)w_frags + (size_t)t * 512 + (size_t)i * 128, 0, 3);
  if (wave == 0) __builtin_amdgcn_s_wait_tensorcnt(0);
  __syncthreads();

  // ---- Phase 1: LayerNorm, 4 rows per wave ----
  for (int r4 = 0; r4 < 4; ++r4) {
    const int row = wave * 4 + r4;
    float x[8], s = 0.f, s2 = 0.f;
#pragma unroll
    for (int i = 0; i < 8; ++i) { x[i] = x_lds[row][i * 32 + lane]; s += x[i]; s2 += x[i] * x[i]; }
#pragma unroll
    for (int off = 16; off > 0; off >>= 1) { s += __shfl_xor(s, off); s2 += __shfl_xor(s2, off); }
    const float mean = s * (1.f / kDF);
    const float rstd = rsqrtf(s2 * (1.f / kDF) - mean * mean + LN_EPS_F);
#pragma unroll
    for (int i = 0; i < 8; ++i) {
      const int d = i * 32 + lane;
      a_lds[row][d] = f2bf((x[i] - mean) * rstd * g_in[d] + b_in[d]);
    }
  }
  __syncthreads();

  // ---- Phase 2: 32 WMMA tile jobs = rt(2) x mat(2) x ct(8), 4 per wave ----
  for (int jj = 0; jj < 4; ++jj) {
    const int job = wave + jj * 8;
    const int rt = job >> 4, mat = (job >> 3) & 1, ct = job & 7;
    v8f acc = {0.f, 0.f, 0.f, 0.f, 0.f, 0.f, 0.f, 0.f};
    const int m = rt * 16 + (lane & 15);
    const int kb = (lane >> 4) * 8;
#pragma unroll
    for (int kk = 0; kk < 8; ++kk) {
      v16bf A;
      unsigned short* ap = (unsigned short*)&A;
#pragma unroll
      for (int j = 0; j < 8; ++j) {
        const int k = kk * 32 + kb + ((j < 4) ? 2 * j : 16 + 2 * (j - 4));
        ap[2 * j]     = a_lds[m][k];
        ap[2 * j + 1] = a_lds[m][k + 1];
      }
      const v16bf Bf = *(const v16bf*)(w_frags + ((((mat * 8 + ct) * 8 + kk) * 32 + lane) << 3));
      acc = __builtin_amdgcn_wmma_f32_16x16x32_bf16(false, A, false, Bf, (short)0, acc, false, false);
    }
    const int dcol = ct * 16 + (lane & 15);
#pragma unroll
    for (int i = 0; i < 8; ++i) {
      const int tokr = rt * 16 + (lane >> 4) * 8 + i;
      kv_stage[mat][tokr][dcol] = f2bf(acc[i]);
    }
  }
  __syncthreads();

  // ---- Phase 3: emit fragments (4096 u32 words / block) ----
#pragma unroll
  for (int w = 0; w < 16; ++w) {
    const int widx = t + (w << 8);
    if (widx < 2048) {  // k-fragments: tt(2) x kt(4)
      const int j = widx & 7, ln = (widx >> 3) & 31, kt = (widx >> 8) & 3, tt = (widx >> 10) & 1;
      const int tokl = tt * 16 + (ln & 15);
      const int d = kt * 32 + (ln >> 4) * 8 + ((j < 4) ? 2 * j : 16 + 2 * (j - 4));
      const unsigned val = (unsigned)kv_stage[0][tokl][d] |
                           ((unsigned)kv_stage[0][tokl][d + 1] << 16);
      k_frags[(((b * 256 + chunk * 2 + tt) * 4 + kt) << 8) + ln * 8 + j] = val;
    } else {            // v-fragments: dt(8)
      const int wv = widx - 2048;
      const int j = wv & 7, ln = (wv >> 3) & 31, dt = (wv >> 8) & 7;
      const int d = dt * 16 + (ln & 15);
      const int tok = (ln >> 4) * 8 + ((j < 4) ? 2 * j : 16 + 2 * (j - 4));
      const unsigned val = (unsigned)kv_stage[1][tok][d] |
                           ((unsigned)kv_stage[1][tok + 1][d] << 16);
      v_frags[(((b * 128 + chunk) * 8 + dt) << 8) + ln * 8 + j] = val;
    }
  }
}

// ---------------------------------------------------------------------------
// q = LN(slots) @ Wq, packed as WMMA A-fragments (16 slots padded, K = d).
// q_frags: [b][kt(4)][lane][8] u32.
// ---------------------------------------------------------------------------
__global__ void slots_q_kernel(const float* __restrict__ slots,
                               const float* __restrict__ g_s,
                               const float* __restrict__ b_s,
                               const float* __restrict__ Wq,
                               unsigned* __restrict__ q_frags) {
  __shared__ float lnq[kS][kDS];
  __shared__ float qv[kS][kDS];
  const int t = threadIdx.x, b = blockIdx.x;
  const int row = t >> 4, sub = t & 15;
  const float* src = slots + b * kS * kDS + row * kDS;
  float x[8], s = 0.f, s2 = 0.f;
#pragma unroll
  for (int i = 0; i < 8; ++i) { x[i] = src[sub + 16 * i]; s += x[i]; s2 += x[i] * x[i]; }
#pragma unroll
  for (int off = 8; off > 0; off >>= 1) { s += __shfl_xor(s, off); s2 += __shfl_xor(s2, off); }
  const float mean = s * (1.f / kDS);
  const float rstd = rsqrtf(s2 * (1.f / kDS) - mean * mean + LN_EPS_F);
#pragma unroll
  for (int i = 0; i < 8; ++i) {
    const int d = sub + 16 * i;
    lnq[row][d] = (x[i] - mean) * rstd * g_s[d] + b_s[d];
  }
  __syncthreads();
#pragma unroll
  for (int o = 0; o < 8; ++o) {
    const int oid = t + o * 128, sx = oid >> 7, cx = oid & 127;
    float a = 0.f;
    for (int d = 0; d < kDS; ++d) a += lnq[sx][d] * Wq[d * kDS + cx];
    qv[sx][cx] = a;
  }
  __syncthreads();
#pragma unroll
  for (int o = 0; o < 8; ++o) {
    const int wid = t + o * 128;
    const int j = wid & 7, ln = (wid >> 3) & 31, kt = (wid >> 8) & 3;
    const int m = ln & 15;
    unsigned val = 0u;
    if (m < 8) {
      const int k = kt * 32 + (ln >> 4) * 8 + ((j < 4) ? 2 * j : 16 + 2 * (j - 4));
      val = pack2(qv[m][k], qv[m][k + 1]);
    }
    q_frags[b * 1024 + wid] = val;
  }
}

// ---------------------------------------------------------------------------
// Fused dots -> softmax(over slots) -> +eps -> attn @ v (unnormalized) with
// per-slot row-sums. WMMA C layout puts all 8 slot values of a token in one
// lane -> softmax over axis=1 is purely in-lane. Grid: (8 splits, B).
// ---------------------------------------------------------------------------
__global__ void attn_update_kernel(const unsigned* __restrict__ q_frags,
                                   const unsigned* __restrict__ k_frags,
                                   const unsigned* __restrict__ v_frags,
                                   float* __restrict__ upd,
                                   float* __restrict__ rowsum) {
  __shared__ unsigned short attn_lds[8][32][16];  // [wave][tok][slot], 8 KB
  __shared__ float upd_sh[16][kDS];               // 8 KB
  const int t = threadIdx.x, wave = t >> 5, lane = t & 31;
  const int b = blockIdx.y, split = blockIdx.x;

#pragma unroll
  for (int i = 0; i < 8; ++i) {
    const int idx = t + i * 256;
    upd_sh[idx >> 7][idx & 127] = 0.f;
  }
  __syncthreads();

  v16bf aq[4];
#pragma unroll
  for (int kt = 0; kt < 4; ++kt)
    aq[kt] = *(const v16bf*)(q_frags + b * 1024 + ((kt * 32 + lane) << 3));

  v8f uacc[8];
#pragma unroll
  for (int dt = 0; dt < 8; ++dt) uacc[dt] = (v8f){0.f, 0.f, 0.f, 0.f, 0.f, 0.f, 0.f, 0.f};
  float rs[8];
#pragma unroll
  for (int i = 0; i < 8; ++i) rs[i] = 0.f;

  const int base = split * 512 + wave * 64;
  for (int c = 0; c < 2; ++c) {
    const int tok0 = base + c * 32;
    const int ch = tok0 >> 5;
#pragma unroll
    for (int st = 0; st < 2; ++st) {
      const int ttile = (tok0 >> 4) + st;
      v8f dacc = {0.f, 0.f, 0.f, 0.f, 0.f, 0.f, 0.f, 0.f};
#pragma unroll
      for (int kt = 0; kt < 4; ++kt) {
        const v16bf Bf =
            *(const v16bf*)(k_frags + ((((b * 256 + ttile) * 4 + kt) << 8) + lane * 8));
        dacc = __builtin_amdgcn_wmma_f32_16x16x32_bf16(false, aq[kt], false, Bf, (short)0,
                                                       dacc, false, false);
      }
      // softmax across the 8 real slots (lanes 0-15 hold them in-register)
      float e[8], mx = -1e30f, se = 0.f;
#pragma unroll
      for (int i = 0; i < 8; ++i) { e[i] = dacc[i] * SCALE_F; mx = fmaxf(mx, e[i]); }
#pragma unroll
      for (int i = 0; i < 8; ++i) { e[i] = __expf(e[i] - mx); se += e[i]; }
      const float inv = 1.f / se;
      const int tk = st * 16 + (lane & 15);
      const int sb = (lane >> 4) * 8;
      if (lane < 16) {
#pragma unroll
        for (int i = 0; i < 8; ++i) {
          const float a = e[i] * inv + EPS_F;
          rs[i] += a;
          attn_lds[wave][tk][i] = f2bf(a);
        }
      } else {  // pad slots 8..15 -> 0
#pragma unroll
        for (int i = 0; i < 8; ++i) attn_lds[wave][tk][sb + i] = 0;
      }
    }
    // attn as A-fragment (16 slots x 32 tokens), then updates WMMAs
    v16bf Af;
    unsigned short* ap = (unsigned short*)&Af;
    const int m = lane & 15, kb = (lane >> 4) * 8;
#pragma unroll
    for (int j = 0; j < 8; ++j) {
      const int k = kb + ((j < 4) ? 2 * j : 16 + 2 * (j - 4));
      ap[2 * j]     = attn_lds[wave][k][m];
      ap[2 * j + 1] = attn_lds[wave][k + 1][m];
    }
#pragma unroll
    for (int dt = 0; dt < 8; ++dt) {
      const v16bf Vf =
          *(const v16bf*)(v_frags + ((((b * 128 + ch) * 8 + dt) << 8) + lane * 8));
      uacc[dt] = __builtin_amdgcn_wmma_f32_16x16x32_bf16(false, Af, false, Vf, (short)0,
                                                         uacc[dt], false, false);
    }
  }

  // row-sum reduce across wave -> one atomic per slot
#pragma unroll
  for (int i = 0; i < 8; ++i) {
#pragma unroll
    for (int off = 16; off > 0; off >>= 1) rs[i] += __shfl_xor(rs[i], off);
  }
  if (lane == 0) {
#pragma unroll
    for (int i = 0; i < 8; ++i) atomicAdd(&rowsum[b * 16 + i], rs[i]);
  }

  // updates: wave partials -> LDS -> single global flush
  const int ncol = lane & 15, mrow = (lane >> 4) * 8;
#pragma unroll
  for (int dt = 0; dt < 8; ++dt) {
#pragma unroll
    for (int i = 0; i < 8; ++i)
      atomicAdd(&upd_sh[mrow + i][dt * 16 + ncol], uacc[dt][i]);
  }
  __syncthreads();
#pragma unroll
  for (int i = 0; i < 8; ++i) {
    const int idx = t + i * 256;
    atomicAdd(&upd[b * 2048 + idx], upd_sh[idx >> 7][idx & 127]);
  }
}

// ---------------------------------------------------------------------------
// updates/rowsum normalization + GRUCell + LN + MLP residual. 8 rows / block.
// ---------------------------------------------------------------------------
__global__ void gru_mlp_kernel(const float* __restrict__ upd,
                               const float* __restrict__ rowsum,
                               const float* __restrict__ slots_prev,
                               const float* __restrict__ W_ih, const float* __restrict__ W_hh,
                               const float* __restrict__ b_ih, const float* __restrict__ b_hh,
                               const float* __restrict__ g_m, const float* __restrict__ b_m,
                               const float* __restrict__ W1, const float* __restrict__ b1,
                               const float* __restrict__ W2, const float* __restrict__ b2,
                               float* __restrict__ out) {
  __shared__ float u_sh[kS][kDS], h_prev[kS][kDS], h_sh[kS][kDS], lnh[kS][kDS];
  __shared__ float gr[kS][kDS], gz[kS][kDS], gnx[kS][kDS], gnh[kS][kDS];
  __shared__ float mid[kS][kMH];
  const int t = threadIdx.x, b = blockIdx.x;

#pragma unroll
  for (int i = 0; i < 4; ++i) {
    const int idx = t + i * 256, s = idx >> 7, d = idx & 127;
    u_sh[s][d] = upd[b * 2048 + idx] / rowsum[b * 16 + s];
    h_prev[s][d] = slots_prev[b * 1024 + idx];
  }
  __syncthreads();

#pragma unroll
  for (int kk = 0; kk < 12; ++kk) {
    const int oid = t + kk * 256, s = oid / 384, g = oid % 384;
    float ax = b_ih[g], ah = b_hh[g];
    for (int d = 0; d < kDS; ++d) {
      ax += u_sh[s][d] * W_ih[g * kDS + d];
      ah += h_prev[s][d] * W_hh[g * kDS + d];
    }
    if (g < 128)      gr[s][g] = ax + ah;
    else if (g < 256) gz[s][g - 128] = ax + ah;
    else            { gnx[s][g - 256] = ax; gnh[s][g - 256] = ah; }
  }
  __syncthreads();

#pragma unroll
  for (int i = 0; i < 4; ++i) {
    const int idx = t + i * 256, s = idx >> 7, d = idx & 127;
    const float r = 1.f / (1.f + __expf(-gr[s][d]));
    const float z = 1.f / (1.f + __expf(-gz[s][d]));
    const float n = tanhf(gnx[s][d] + r * gnh[s][d]);
    h_sh[s][d] = (1.f - z) * n + z * h_prev[s][d];
  }
  __syncthreads();

  {  // LayerNorm(h), one wave per row
    const int row = t >> 5, l = t & 31;
    float s0 = 0.f, s1 = 0.f, xv[4];
#pragma unroll
    for (int i = 0; i < 4; ++i) { xv[i] = h_sh[row][l + 32 * i]; s0 += xv[i]; s1 += xv[i] * xv[i]; }
#pragma unroll
    for (int off = 16; off > 0; off >>= 1) { s0 += __shfl_xor(s0, off); s1 += __shfl_xor(s1, off); }
    const float mean = s0 * (1.f / kDS);
    const float rstd = rsqrtf(s1 * (1.f / kDS) - mean * mean + LN_EPS_F);
#pragma unroll
    for (int i = 0; i < 4; ++i) {
      const int d = l + 32 * i;
      lnh[row][d] = (xv[i] - mean) * rstd * g_m[d] + b_m[d];
    }
  }
  __syncthreads();

#pragma unroll
  for (int kk = 0; kk < 8; ++kk) {
    const int oid = t + kk * 256, s = oid >> 8, c = oid & 255;
    float a = b1[c];
    for (int d = 0; d < kDS; ++d) a += lnh[s][d] * W1[d * kMH + c];
    mid[s][c] = fmaxf(a, 0.f);
  }
  __syncthreads();

#pragma unroll
  for (int i = 0; i < 4; ++i) {
    const int idx = t + i * 256, s = idx >> 7, d = idx & 127;
    float a = b2[d];
    for (int c = 0; c < kMH; ++c) a += mid[s][c] * W2[c * kDS + d];
    out[b * 1024 + idx] = h_sh[s][d] + a;
  }
}

// ---------------------------------------------------------------------------
extern "C" void kernel_launch(void* const* d_in, const int* in_sizes, int n_in,
                              void* d_out, int out_size, void* d_ws, size_t ws_size,
                              hipStream_t stream) {
  (void)in_sizes; (void)n_in; (void)out_size; (void)ws_size;
  const float* inputs = (const float*)d_in[0];
  const float* slots  = (const float*)d_in[1];
  const float* g_in = (const float*)d_in[2];
  const float* b_in = (const float*)d_in[3];
  const float* g_s  = (const float*)d_in[4];
  const float* b_s  = (const float*)d_in[5];
  const float* g_m  = (const float*)d_in[6];
  const float* b_m  = (const float*)d_in[7];
  const float* Wq   = (const float*)d_in[8];
  const float* Wk   = (const float*)d_in[9];
  const float* Wv   = (const float*)d_in[10];
  const float* W_ih = (const float*)d_in[11];
  const float* W_hh = (const float*)d_in[12];
  const float* b_ih = (const float*)d_in[13];
  const float* b_hh = (const float*)d_in[14];
  const float* W1   = (const float*)d_in[15];
  const float* b1   = (const float*)d_in[16];
  const float* W2   = (const float*)d_in[17];
  const float* b2   = (const float*)d_in[18];
  // d_in[19] = step (int), unused: both iter counts are 3.

  char* ws = (char*)d_ws;
  unsigned* k_frags = (unsigned*)(ws);                               // 64 MiB
  unsigned* v_frags = (unsigned*)(ws + (size_t)67108864);            // 64 MiB
  unsigned* w_frags = (unsigned*)(ws + (size_t)134217728);           // 128 KiB
  unsigned* q_frags = (unsigned*)(ws + (size_t)134217728 + 131072);  // 256 KiB
  float* upd    = (float*)(ws + (size_t)134217728 + 131072 + 262144);
  float* rowsum = (float*)((char*)upd + 524288);
  float* slotsA = (float*)((char*)rowsum + 4096);
  float* slotsB = (float*)((char*)slotsA + 262144);

  prep_wfrags_kernel<<<128, 256, 0, stream>>>(Wk, Wv, w_frags);
  ln_kv_kernel<<<kB * kN / 32, 256, 0, stream>>>(inputs, g_in, b_in, w_frags, k_frags, v_frags);

  const float* cur = slots;
  float* outs[3] = {slotsA, slotsB, (float*)d_out};
  for (int it = 0; it < 3; ++it) {
    slots_q_kernel<<<kB, 128, 0, stream>>>(cur, g_s, b_s, Wq, q_frags);
    (void)hipMemsetAsync(upd, 0, (size_t)kB * 16 * kDS * sizeof(float), stream);
    (void)hipMemsetAsync(rowsum, 0, (size_t)kB * 16 * sizeof(float), stream);
    attn_update_kernel<<<dim3(8, kB), 256, 0, stream>>>(q_frags, k_frags, v_frags, upd, rowsum);
    gru_mlp_kernel<<<kB, 256, 0, stream>>>(upd, rowsum, cur, W_ih, W_hh, b_ih, b_hh,
                                           g_m, b_m, W1, b1, W2, b2, outs[it]);
    cur = outs[it];
  }
}
